// TimeAwareLSTM_35888746725955
// MI455X (gfx1250) — matrix operand order repaired and checked by reading, not compiled, not run on hardware
//
#include <hip/hip_runtime.h>

#define INDIM 256
#define HS    512
#define SEQ   512
#define NB    256
#define MB    16      // batch rows per workgroup (== WMMA M tile)
#define BLOCK 1024
#define NWAVES 32
#define KTX   8       // k-tiles from x:  256/32
#define KTH   16      // k-tiles from h:  512/32
#define KTG   24      // total gate k-tiles
#define KTT   16      // k-tiles for c@Wt gemm: 512/32
#define NTG   128     // n-tiles for gates: 2048/16
#define NTT   32      // n-tiles for Wt:    512/16
#define NROWS ((size_t)NB * SEQ)          // 131072 flattened (b,s) rows

typedef __attribute__((ext_vector_type(16))) __bf16 v16bf;
typedef __attribute__((ext_vector_type(8)))  float  v8f;

union Frag { uint4 q[2]; unsigned short s[16]; v16bf v; };

__device__ __forceinline__ unsigned short f2bf(float f) {
  unsigned u = __builtin_bit_cast(unsigned, f);
  u += 0x7FFFu + ((u >> 16) & 1u);          // round-to-nearest-even
  return (unsigned short)(u >> 16);
}
__device__ __forceinline__ float bf2f(unsigned short s) {
  unsigned u = ((unsigned)s) << 16;
  return __builtin_bit_cast(float, u);
}
// fast transcendentals: v_exp_f32 + v_rcp_f32, clamped for safety
__device__ __forceinline__ float fast_tanh(float x) {
  float cx = fminf(fmaxf(x, -15.0f), 15.0f);
  float e  = __expf(2.0f * cx);
  return (e - 1.0f) * __builtin_amdgcn_rcpf(e + 1.0f);
}
__device__ __forceinline__ float fast_sigmoid(float x) {
  return __builtin_amdgcn_rcpf(1.0f + __expf(-x));
}
__device__ __forceinline__ v8f wmma_bf16(v16bf a, v16bf b, v8f c) {
  return __builtin_amdgcn_wmma_f32_16x16x32_bf16(false, a, false, b, (short)0, c,
                                                 false, false);
}
// B fragment: packed lane-major tile, 32 contiguous bytes per lane
__device__ __forceinline__ v16bf load_b_frag(const unsigned short* tile, int lane) {
  Frag f;
  const uint4* p = (const uint4*)(tile + lane * 16);
  f.q[0] = p[0]; f.q[1] = p[1];
  return f.v;
}
// A fragment from bf16 LDS: lanes 0-15: K = e | e+8 split; lanes 16-31: +8 (ISA 7.12.2)
__device__ __forceinline__ v16bf load_a_bf(const unsigned short* base, int stride,
                                           int m, int k0) {
  Frag f;
  f.q[0] = *(const uint4*)(base + m * stride + k0);
  f.q[1] = *(const uint4*)(base + m * stride + k0 + 16);
  return f.v;
}

// One-shot repack: [Wx;Wh] (768x2048) and Wt (512x512) f32 -> bf16 WMMA-B tiles.
__global__ void talstm_pack_weights(const float* __restrict__ wx,
                                    const float* __restrict__ wh,
                                    const float* __restrict__ wt,
                                    unsigned short* __restrict__ wsg,
                                    unsigned short* __restrict__ wst) {
  int tid = blockIdx.x * blockDim.x + threadIdx.x;
  const int EG = NTG * KTG * 512;
  const int ET = NTT * KTT * 512;
  if (tid < EG) {
    int t = tid >> 9, off = tid & 511;
    int lane = off >> 4, e = off & 15;
    int n_in = lane & 15, k_in = (e & 7) + ((lane >> 4) << 3) + ((e >> 3) << 4);
    int nt = t / KTG, kt = t % KTG;
    int K = kt * 32 + k_in, N = nt * 16 + n_in;
    float v = (K < INDIM) ? wx[K * 2048 + N] : wh[(K - INDIM) * 2048 + N];
    wsg[tid] = f2bf(v);
  } else if (tid < EG + ET) {
    int id = tid - EG;
    int t = id >> 9, off = id & 511;
    int lane = off >> 4, e = off & 15;
    int n_in = lane & 15, k_in = (e & 7) + ((lane >> 4) << 3) + ((e >> 3) << 4);
    int nt = t >> 4, kt = t & 15;
    int K = kt * 32 + k_in, N = nt * 16 + n_in;
    wst[id] = f2bf(wt[K * 512 + N]);
  }
}

// Fully parallel pre-GEMM: x_proj = x @ Wx + b_g  (131072x256 @ 256x2048).
// 262144 independent waves saturate the chip; removes 8 of 24 k-tiles (and all
// x staging/conversion) from the 512-step serial recurrence.
__global__ __launch_bounds__(256) void talstm_xproj(
    const float* __restrict__ x, const float* __restrict__ bias,
    const unsigned short* __restrict__ wg, float* __restrict__ xp) {
  int gw   = (blockIdx.x * 256 + threadIdx.x) >> 5;   // wave id 0..262143
  int lane = threadIdx.x & 31;
  int hi = lane >> 4, mm = lane & 15;
  int ntq = gw & 31;            // group of 4 n-tiles
  int mt  = gw >> 5;            // row tile 0..8191
  int row = mt * 16 + mm;
  const float* xr = x + (size_t)row * INDIM;
  const int k0a = 8 * hi;

  v8f acc[4];
  #pragma unroll
  for (int j = 0; j < 4; ++j) {
    float bv = bias[HS + (ntq * 4 + j) * 16 + mm];
    #pragma unroll
    for (int r = 0; r < 8; ++r) acc[j][r] = bv;
  }
  for (int kk = 0; kk < KTX; ++kk) {
    Frag fa;
    #pragma unroll
    for (int i = 0; i < 8; ++i) fa.s[i]     = f2bf(xr[kk * 32 + k0a + i]);
    #pragma unroll
    for (int i = 0; i < 8; ++i) fa.s[8 + i] = f2bf(xr[kk * 32 + k0a + 16 + i]);
    #pragma unroll
    for (int j = 0; j < 4; ++j) {
      int nt = ntq * 4 + j;
      v16bf b = load_b_frag(wg + (size_t)(nt * KTG + kk) * 512, lane);
      acc[j] = wmma_bf16(fa.v, b, acc[j]);
    }
  }
  #pragma unroll
  for (int j = 0; j < 4; ++j)
    #pragma unroll
    for (int r = 0; r < 8; ++r)
      xp[(size_t)(mt * 16 + r + 8 * hi) * 2048 + (ntq * 4 + j) * 16 + mm] = acc[j][r];
}

// Persistent recurrent kernel: 16 WGs x 32 waves; each WG owns 16 batch rows.
// f32 cell state in REGISTERS; bf16 shadows of c,h in LDS feed the WMMA A-side.
// USE_XP: gate accumulators initialized from precomputed x_proj (K drops to 512).
template <bool USE_XP>
__global__ __launch_bounds__(BLOCK) void talstm_kernel(
    const float* __restrict__ x, const float* __restrict__ timediff,
    const float* __restrict__ bias,
    const unsigned short* __restrict__ wg,   // packed [128 nt][24 kt][512]
    const unsigned short* __restrict__ wt,   // packed [ 32 nt][16 kt][512]
    const float* __restrict__ xp,            // x_proj (or nullptr)
    float* __restrict__ out) {
  __shared__ unsigned short lds_cb[MB * HS];     // cell state (bf16)   16 KB
  __shared__ unsigned short lds_h[MB * HS];      // hidden (bf16)       16 KB
  __shared__ unsigned short lds_x[MB * INDIM];   // x_t (bf16, fused path)
  __shared__ float          lds_ttd[MB];         // tanh(TimeDiff) per row

  const int tid  = threadIdx.x;
  const int w    = tid >> 5;
  const int lane = tid & 31;
  const int hi   = lane >> 4;
  const int m    = lane & 15;
  const int b0   = blockIdx.x * MB;
  const int n    = w * 16 + m;     // owned hidden column
  const int k0a  = 8 * hi;

  // wave-uniform SGPR stream offsets; pointer chain rooted at kernel args so
  // addrspace(1) inference survives -> global_load (not flat).
  const unsigned wt_off = __builtin_amdgcn_readfirstlane(w * (KTT * 512));
  const unsigned wg_off = __builtin_amdgcn_readfirstlane(w * (KTG * 512));
  const size_t GSTR = (size_t)NWAVES * KTG * 512;

  for (int i = tid; i < MB * HS; i += BLOCK) { lds_cb[i] = 0; lds_h[i] = 0; }
  __syncthreads();

  const float bt = bias[n];
  float bg[4];
  #pragma unroll
  for (int g = 0; g < 4; ++g) bg[g] = USE_XP ? 0.0f : bias[HS + g * HS + n];

  v8f c_reg;
  #pragma unroll
  for (int r = 0; r < 8; ++r) c_reg[r] = 0.0f;

  for (int t = 0; t < SEQ; ++t) {
    // opaque uniform zero per iteration: blocks LICM hoist-and-spill of the
    // t-invariant weight stream while keeping pointer provenance intact.
    unsigned zz = 0;
    asm volatile("" : "+s"(zz));
    const unsigned short* wtb_t = wt + wt_off + zz;
    const unsigned short* wgb_t = wg + wg_off + zz;

    // ---- stage tanh(td); fused path also stages x_t -> bf16 LDS ----
    if (!USE_XP && tid < 512) {
      int row = tid >> 5;
      int c8  = (tid & 31) * 8;
      const float* src = x + ((size_t)(b0 + row) * SEQ + t) * INDIM + c8;
      float4 f0 = *(const float4*)(src);
      float4 f1 = *(const float4*)(src + 4);
      uint4 pk;
      pk.x = (unsigned)f2bf(f0.x) | ((unsigned)f2bf(f0.y) << 16);
      pk.y = (unsigned)f2bf(f0.z) | ((unsigned)f2bf(f0.w) << 16);
      pk.z = (unsigned)f2bf(f1.x) | ((unsigned)f2bf(f1.y) << 16);
      pk.w = (unsigned)f2bf(f1.z) | ((unsigned)f2bf(f1.w) << 16);
      *(uint4*)(lds_x + row * INDIM + c8) = pk;
    }
    if (tid < MB) lds_ttd[tid] = fast_tanh(timediff[(size_t)(b0 + tid) * SEQ + t]);
    __syncthreads();

    // ---- c_s pre-activation = c @ Wt (+ b_t inside tanh)  (K = 512) ----
    v8f acc_cs;
    #pragma unroll
    for (int r = 0; r < 8; ++r) acc_cs[r] = 0.0f;
    for (int kk = 0; kk < KTT; ++kk) {
      v16bf a = load_a_bf(lds_cb, HS, m, kk * 32 + k0a);
      v16bf b = load_b_frag(wtb_t + ((size_t)kk << 9), lane);
      acc_cs = wmma_bf16(a, b, acc_cs);
    }

    // ---- c_star = c - c_s*(1 + tanh(td))   (pure register math) ----
    v8f cstar;
    #pragma unroll
    for (int r = 0; r < 8; ++r) {
      float s = fast_tanh(acc_cs[r] + bt);
      cstar[r] = c_reg[r] - s * (1.0f + lds_ttd[r + 8 * hi]);
    }

    // ---- gates: acc init from x_proj (XP) or zero; then h @ Wh ----
    v8f acc[4];
    if (USE_XP) {
      #pragma unroll
      for (int g = 0; g < 4; ++g)
        #pragma unroll
        for (int r = 0; r < 8; ++r)
          acc[g][r] = xp[((size_t)(b0 + r + 8 * hi) * SEQ + t) * 2048 + g * HS + n];
    } else {
      #pragma unroll
      for (int g = 0; g < 4; ++g)
        #pragma unroll
        for (int r = 0; r < 8; ++r) acc[g][r] = 0.0f;
      for (int kk = 0; kk < KTX; ++kk) {
        v16bf a = load_a_bf(lds_x, INDIM, m, kk * 32 + k0a);
        const unsigned short* bp = wgb_t + ((size_t)kk << 9);
        #pragma unroll
        for (int g = 0; g < 4; ++g)
          acc[g] = wmma_bf16(a, load_b_frag(bp + (size_t)g * GSTR, lane), acc[g]);
      }
    }
    for (int kk = 0; kk < KTH; ++kk) {
      v16bf a = load_a_bf(lds_h, HS, m, kk * 32 + k0a);
      const unsigned short* bp = wgb_t + ((size_t)(KTX + kk) << 9);
      #pragma unroll
      for (int g = 0; g < 4; ++g)
        acc[g] = wmma_bf16(a, load_b_frag(bp + (size_t)g * GSTR, lane), acc[g]);
    }
    __syncthreads();   // all waves done reading lds_cb / lds_h / lds_x

    // ---- elementwise combine + outputs (owned columns) ----
    #pragma unroll
    for (int r = 0; r < 8; ++r) {
      int M = r + 8 * hi;
      float iv = fast_sigmoid(acc[0][r] + bg[0]);
      float fv = fast_sigmoid(acc[1][r] + bg[1]);
      float ov = fast_sigmoid(acc[2][r] + bg[2]);
      float cv = fast_tanh(acc[3][r] + bg[3]);
      float cn = fv * cstar[r] + iv * cv;
      float hn = ov + fast_tanh(cn);       // reference: h = o_t + tanh(c_new)
      c_reg[r] = cn;
      lds_cb[M * HS + n] = f2bf(cn);
      lds_h[M * HS + n]  = f2bf(hn);
      __builtin_nontemporal_store(hn, &out[((size_t)(b0 + M) * SEQ + t) * HS + n]);
    }
  }
  __syncthreads();

  // ---- final (h_T, c_T) ----
  float* hT = out + (size_t)NB * SEQ * HS;
  float* cT = hT + (size_t)NB * HS;
  #pragma unroll
  for (int r = 0; r < 8; ++r) {
    int M = r + 8 * hi;
    cT[(size_t)(b0 + M) * HS + n] = c_reg[r];
  }
  for (int i = tid; i < MB * HS; i += BLOCK) {
    int mm = i >> 9, nn = i & (HS - 1);
    hT[(size_t)(b0 + mm) * HS + nn] = bf2f(lds_h[i]);
  }
}

extern "C" void kernel_launch(void* const* d_in, const int* in_sizes, int n_in,
                              void* d_out, int out_size, void* d_ws, size_t ws_size,
                              hipStream_t stream) {
  const float* x    = (const float*)d_in[0];
  const float* td   = (const float*)d_in[1];
  const float* wx   = (const float*)d_in[2];
  const float* wh   = (const float*)d_in[3];
  const float* wt   = (const float*)d_in[4];
  const float* bias = (const float*)d_in[5];

  unsigned short* wsg = (unsigned short*)d_ws;               // 3 MB packed gates W
  unsigned short* wst = wsg + (size_t)NTG * KTG * 512;       // 0.5 MB packed Wt
  float* xproj = (float*)(wst + (size_t)NTT * KTT * 512);    // 1 GB x_proj (optional)

  size_t packed_bytes = ((size_t)NTG * KTG * 512 + (size_t)NTT * KTT * 512) * 2;
  size_t xp_bytes     = NROWS * 2048 * sizeof(float);
  bool use_xp = ws_size >= packed_bytes + xp_bytes;

  int total = NTG * KTG * 512 + NTT * KTT * 512;
  talstm_pack_weights<<<(total + 255) / 256, 256, 0, stream>>>(wx, wh, wt, wsg, wst);

  if (use_xp) {
    talstm_xproj<<<32768, 256, 0, stream>>>(x, bias, wsg, xproj);
    talstm_kernel<true><<<NB / MB, BLOCK, 0, stream>>>(x, td, bias, wsg, wst, xproj,
                                                       (float*)d_out);
  } else {
    talstm_kernel<false><<<NB / MB, BLOCK, 0, stream>>>(x, td, bias, wsg, wst, nullptr,
                                                        (float*)d_out);
  }
}
